// MultiHeadedAttention_80814104642387
// MI455X (gfx1250) — matrix operand order repaired and checked by reading, not compiled, not run on hardware
//
#include <hip/hip_runtime.h>
#include <hip/hip_bf16.h>

// ---------- vector types ----------
typedef _Float16 f16x16 __attribute__((ext_vector_type(16)));
typedef _Float16 f16x8  __attribute__((ext_vector_type(8)));
typedef float    f32x8  __attribute__((ext_vector_type(8)));
typedef float    f32x4  __attribute__((ext_vector_type(4)));
typedef int      i32x4v __attribute__((__vector_size__(4 * sizeof(int))));

#define Bb   4
#define Ss   2048
#define Dd   1024
#define Hh   16
#define DPH  64

#define AS1 __attribute__((address_space(1)))
#define AS3 __attribute__((address_space(3)))

static __device__ inline f32x8 wmma_f16(f16x16 a, f16x16 b, f32x8 c) {
  return __builtin_amdgcn_wmma_f32_16x16x32_f16(false, a, false, b, (short)0, c,
                                                false, false);
}

static __device__ inline f32x8 splat8(float x) {
  f32x8 v;
#pragma unroll
  for (int i = 0; i < 8; ++i) v[i] = x;
  return v;
}

// ---- async global->LDS copy (16B), ASYNCcnt-tracked on CDNA5 ----
static __device__ inline void async_copy16(const _Float16* g, _Float16* l) {
#if __has_builtin(__builtin_amdgcn_global_load_async_to_lds_b128)
  __builtin_amdgcn_global_load_async_to_lds_b128(
      (AS1 i32x4v*)g, (AS3 i32x4v*)l, 0, 0);
#else
  *(f16x8*)l = *(const f16x8*)g;
#endif
}

static __device__ inline void wait_async() {
#if __has_builtin(__builtin_amdgcn_global_load_async_to_lds_b128)
#if __has_builtin(__builtin_amdgcn_s_wait_asynccnt)
  __builtin_amdgcn_s_wait_asynccnt(0);
#else
  asm volatile("s_wait_asynccnt 0" ::: "memory");
#endif
#endif
}

// =====================================================================
// Pre-pass A: f32 -> f16 weight conversion
// =====================================================================
__global__ __launch_bounds__(256) void cvt_f16_kernel(const float* __restrict__ src,
                                                      _Float16* __restrict__ dst) {
  const size_t i = ((size_t)blockIdx.x * 256 + threadIdx.x) * 8;
  const f32x4 a = *(const f32x4*)(src + i);
  const f32x4 b = *(const f32x4*)(src + i + 4);
  f16x8 o;
#pragma unroll
  for (int j = 0; j < 4; ++j) { o[j] = (_Float16)a[j]; o[4 + j] = (_Float16)b[j]; }
  *(f16x8*)(dst + i) = o;
}

// =====================================================================
// Pre-pass B: pack bool mask into bitmask words (u32 covers 32 keys)
// =====================================================================
__global__ __launch_bounds__(256) void mask_pack_kernel(const unsigned char* __restrict__ mask,
                                                        unsigned int* __restrict__ bits) {
  const size_t w = (size_t)blockIdx.x * 256 + threadIdx.x;
  const unsigned int* ms = (const unsigned int*)(mask + w * 32);
  unsigned int out = 0;
#pragma unroll
  for (int d = 0; d < 8; ++d) {
    const unsigned int x = ms[d];
    out |= ((x & 1u) << (4 * d)) | (((x >> 8) & 1u) << (4 * d + 1)) |
           (((x >> 16) & 1u) << (4 * d + 2)) | (((x >> 24) & 1u) << (4 * d + 3));
  }
  bits[w] = out;
}

// =====================================================================
// Stage 1/3 GEMM, double-buffered LDS weight tile.
// MODE 0: A f32, out f16 -> [B,H,S,DPH]          (Q scale=1/8, V)
// MODE 2: A f32, out f16 -> [B,H,DPH,S] (transposed, packed b128 stores)
// MODE 3: A f16, out f32 -> d_out [B,S,D]
// =====================================================================
template <int MODE>
__global__ __launch_bounds__(256) void gemm_kernel(const void* __restrict__ Ain,
                                                   const _Float16* __restrict__ W16,
                                                   const float* __restrict__ bias,
                                                   void* __restrict__ Out,
                                                   float scale) {
  __shared__ _Float16 sW[2][32 * 64];  // ping-pong, 2 x 4KB

  const int tid  = threadIdx.x;
  const int wave = tid >> 5;
  const int lane = tid & 31;
  const int m  = lane & 15;
  const int hh = lane >> 4;
  const int rowBase = blockIdx.x * 128 + wave * 16;
  const int colBase = blockIdx.y * 64;
  const int row = rowBase + m;

  const _Float16* wsrc = W16 + (size_t)(tid >> 3) * Dd + colBase + (tid & 7) * 8;
  const int wlds = (tid >> 3) * 64 + (tid & 7) * 8;

  f32x8 acc[4];
#pragma unroll
  for (int t = 0; t < 4; ++t) acc[t] = splat8(0.0f);

  // preload tile 0
  async_copy16(wsrc, &sW[0][wlds]);
  wait_async();
  __syncthreads();

  for (int kk = 0; kk < Dd; kk += 32) {
    const int buf = (kk >> 5) & 1;
    // issue next weight tile while computing on current one
    if (kk + 32 < Dd)
      async_copy16(wsrc + (size_t)(kk + 32) * Dd, &sW[buf ^ 1][wlds]);
    if (kk + 64 < Dd)
      __builtin_prefetch(wsrc + (size_t)(kk + 64) * Dd);

    // ---- A fragment: 16x32 ----
    f16x16 a;
    if constexpr (MODE == 3) {
      const _Float16* A = (const _Float16*)Ain;
      const f16x8 lo = *(const f16x8*)(A + (size_t)row * Dd + kk + hh * 8);
      const f16x8 hi = *(const f16x8*)(A + (size_t)row * Dd + kk + 16 + hh * 8);
#pragma unroll
      for (int j = 0; j < 8; ++j) { a[j] = lo[j]; a[8 + j] = hi[j]; }
    } else {
      const float* A = (const float*)Ain;
      const float* ap = A + (size_t)row * Dd + kk;
      const f32x4 f0 = *(const f32x4*)(ap + hh * 8);
      const f32x4 f1 = *(const f32x4*)(ap + hh * 8 + 4);
      const f32x4 f2 = *(const f32x4*)(ap + 16 + hh * 8);
      const f32x4 f3 = *(const f32x4*)(ap + 16 + hh * 8 + 4);
#pragma unroll
      for (int j = 0; j < 4; ++j) {
        a[j]      = (_Float16)f0[j];
        a[4 + j]  = (_Float16)f1[j];
        a[8 + j]  = (_Float16)f2[j];
        a[12 + j] = (_Float16)f3[j];
      }
    }

    // ---- B fragments from LDS: lane = N col, K rows in table order ----
#pragma unroll
    for (int t = 0; t < 4; ++t) {
      const _Float16* wp = &sW[buf][t * 16 + m];
      f16x16 bf;
#pragma unroll
      for (int j = 0; j < 8; ++j) { bf[j] = wp[(hh * 8 + j) * 64]; bf[8 + j] = wp[(16 + hh * 8 + j) * 64]; }
      acc[t] = wmma_f16(a, bf, acc[t]);
    }

    wait_async();     // next tile has landed (latency hidden by the math above)
    __syncthreads();
  }

  // ---- epilogue ----
#pragma unroll
  for (int t = 0; t < 4; ++t) {
    const int col = colBase + t * 16 + (lane & 15);
    const float bval = bias[col];
    if constexpr (MODE == 2) {
      const int bidx = rowBase >> 11;
      const int s0b = (rowBase & 2047) + 8 * hh;
      const int head = col >> 6, dph = col & 63;
      f16x8 pk;
#pragma unroll
      for (int r = 0; r < 8; ++r) pk[r] = (_Float16)((acc[t][r] + bval) * scale);
      _Float16* O = (_Float16*)Out;
      *(f16x8*)(O + (((size_t)bidx * Hh + head) * DPH + dph) * Ss + s0b) = pk;
    } else {
#pragma unroll
      for (int r = 0; r < 8; ++r) {
        const int ro = rowBase + r + 8 * hh;
        const float val = (acc[t][r] + bval) * scale;
        if constexpr (MODE == 3) {
          ((float*)Out)[(size_t)ro * Dd + col] = val;
        } else {
          const int bidx = ro >> 11, s = ro & 2047;
          const int head = col >> 6, dph = col & 63;
          ((_Float16*)Out)[(((size_t)bidx * Hh + head) * Ss + s) * DPH + dph] = (_Float16)val;
        }
      }
    }
  }
}

// =====================================================================
// Stage 2: flash attention; double-buffered async K/V staging; softmax
// denominator accumulated by WMMA against an all-ones B matrix.
// =====================================================================
__global__ __launch_bounds__(256) void attn_kernel(
    const _Float16* __restrict__ q16,        // [B,H,S,DPH] (pre-scaled 1/8)
    const _Float16* __restrict__ k16,        // [B,H,DPH,S]
    const _Float16* __restrict__ v16,        // [B,H,S,DPH]
    const unsigned int* __restrict__ mbits,  // [B,S,S/32]
    _Float16* __restrict__ ctx16) {          // [B,S,H,DPH]
  __shared__ _Float16 sK[2][DPH * 32];  // [d][key]  ping-pong
  __shared__ _Float16 sV[2][32 * DPH];  // [key][d]  ping-pong
  __shared__ _Float16 sP[8][16 * 32];   // per-wave P strip

  const int tid  = threadIdx.x;
  const int wave = tid >> 5;
  const int lane = tid & 31;
  const int m  = lane & 15;
  const int hh = lane >> 4;
  const int bh = blockIdx.y;
  const int b  = bh >> 4;
  const int head = bh & 15;
  const int qw = blockIdx.x * 128 + wave * 16;

  // resident Q fragments
  f16x16 qa[2];
  {
    const _Float16* qp = q16 + ((size_t)bh * Ss + (qw + m)) * DPH;
#pragma unroll
    for (int dt = 0; dt < 2; ++dt) {
      const f16x8 lo = *(const f16x8*)(qp + dt * 32 + hh * 8);
      const f16x8 hi = *(const f16x8*)(qp + dt * 32 + 16 + hh * 8);
#pragma unroll
      for (int j = 0; j < 8; ++j) { qa[dt][j] = lo[j]; qa[dt][8 + j] = hi[j]; }
    }
  }

  f16x16 ones;
#pragma unroll
  for (int j = 0; j < 16; ++j) ones[j] = (_Float16)1.0f;

  f32x8 cacc[4];
#pragma unroll
  for (int t = 0; t < 4; ++t) cacc[t] = splat8(0.0f);
  f32x8 lacc = splat8(0.0f);       // softmax denominator, via P @ ones
  f32x8 mrow = splat8(-3.0e38f);

  const _Float16* kbase = k16 + ((size_t)bh * DPH + (tid >> 2)) * Ss + (tid & 3) * 8;
  const _Float16* vbase = v16 + ((size_t)bh * Ss + (tid >> 3)) * DPH + (tid & 7) * 8;
  const int klds = (tid >> 2) * 32 + (tid & 3) * 8;
  const int vlds = (tid >> 3) * DPH + (tid & 7) * 8;

  // preload block 0
  async_copy16(kbase, &sK[0][klds]);
  async_copy16(vbase, &sV[0][vlds]);
  wait_async();
  __syncthreads();

  for (int kb = 0; kb < Ss; kb += 32) {
    const int buf = (kb >> 5) & 1;
    if (kb + 32 < Ss) {
      async_copy16(kbase + kb + 32, &sK[buf ^ 1][klds]);
      async_copy16(vbase + (size_t)(kb + 32) * DPH, &sV[buf ^ 1][vlds]);
    }

    // ---- scores ----
    f32x8 s0, s1;
    {
      f16x16 bk;
#pragma unroll
      for (int nt = 0; nt < 2; ++nt) {
        f32x8 sacc = splat8(0.0f);
#pragma unroll
        for (int dt = 0; dt < 2; ++dt) {
          const _Float16* kp = &sK[buf][(dt * 32) * 32 + nt * 16 + m];
#pragma unroll
          for (int j = 0; j < 8; ++j) { bk[j] = kp[(hh * 8 + j) * 32]; bk[8 + j] = kp[(16 + hh * 8 + j) * 32]; }
          sacc = wmma_f16(qa[dt], bk, sacc);
        }
        if (nt == 0) s0 = sacc; else s1 = sacc;
      }
    }

    // ---- mask via packed bits ----
#pragma unroll
    for (int r = 0; r < 8; ++r) {
      const int qrow = qw + r + 8 * hh;
      const unsigned int word =
          mbits[((size_t)b * Ss + qrow) * (Ss / 32) + (kb >> 5)];
      if (word & (1u << (lane & 15)))        s0[r] = -1.0e18f;
      if (word & (1u << ((lane & 15) + 16))) s1[r] = -1.0e18f;
    }

    // ---- online softmax: row max via shuffles; sum via WMMA later ----
    f32x8 c;
#pragma unroll
    for (int r = 0; r < 8; ++r) {
      float v = fmaxf(s0[r], s1[r]);
#pragma unroll
      for (int off = 1; off < 16; off <<= 1) v = fmaxf(v, __shfl_xor(v, off, 32));
      const float mn = fmaxf(mrow[r], v);
      c[r] = __expf(mrow[r] - mn);
      s0[r] = __expf(s0[r] - mn);
      s1[r] = __expf(s1[r] - mn);
      mrow[r] = mn;
    }
#pragma unroll
    for (int t = 0; t < 4; ++t)
#pragma unroll
      for (int r = 0; r < 8; ++r) cacc[t][r] *= c[r];
#pragma unroll
    for (int r = 0; r < 8; ++r) lacc[r] *= c[r];

    // ---- P: C-layout -> A-layout via per-wave LDS strip ----
    {
      _Float16* pw = sP[wave];
#pragma unroll
      for (int r = 0; r < 8; ++r) {
        const int prow = r + 8 * hh;
        pw[prow * 32 + (lane & 15)]      = (_Float16)s0[r];
        pw[prow * 32 + 16 + (lane & 15)] = (_Float16)s1[r];
      }
    }
    asm volatile("" ::: "memory");  // LDS in-order per wave; pin compiler order
    f16x16 pa;
    {
      const _Float16* pw = sP[wave] + m * 32;
      const f16x8 lo = *(const f16x8*)(pw + hh * 8);
      const f16x8 hi = *(const f16x8*)(pw + 16 + hh * 8);
#pragma unroll
      for (int j = 0; j < 8; ++j) { pa[j] = lo[j]; pa[8 + j] = hi[j]; }
    }

    // ---- denominator and context, all on the matrix pipe ----
    lacc = wmma_f16(pa, ones, lacc);
#pragma unroll
    for (int dt = 0; dt < 4; ++dt) {
      const _Float16* vp = &sV[buf][dt * 16 + m];
      f16x16 bv;
#pragma unroll
      for (int j = 0; j < 8; ++j) { bv[j] = vp[(hh * 8 + j) * DPH]; bv[8 + j] = vp[(16 + hh * 8 + j) * DPH]; }
      cacc[dt] = wmma_f16(pa, bv, cacc[dt]);
    }

    wait_async();     // next K/V tiles landed while we did the math
    __syncthreads();
  }

  // ---- normalize, store f16 [B,S,H,DPH] ----
#pragma unroll
  for (int dt = 0; dt < 4; ++dt) {
    const int dph = dt * 16 + (lane & 15);
#pragma unroll
    for (int r = 0; r < 8; ++r) {
      const int s = qw + r + 8 * hh;
      const float val = cacc[dt][r] / lacc[r];
      ctx16[(((size_t)b * Ss + s) * Hh + head) * DPH + dph] = (_Float16)val;
    }
  }
}

// =====================================================================
// Launch
// =====================================================================
extern "C" void kernel_launch(void* const* d_in, const int* in_sizes, int n_in,
                              void* d_out, int out_size, void* d_ws, size_t ws_size,
                              hipStream_t stream) {
  const float* key   = (const float*)d_in[0];
  const float* value = (const float*)d_in[1];
  const float* query = (const float*)d_in[2];
  const unsigned char* mask = (const unsigned char*)d_in[3];
  const float* Wq = (const float*)d_in[4];
  const float* bq = (const float*)d_in[5];
  const float* Wk = (const float*)d_in[6];
  const float* bk = (const float*)d_in[7];
  const float* Wv = (const float*)d_in[8];
  const float* bv = (const float*)d_in[9];
  const float* Wo = (const float*)d_in[10];
  const float* bo = (const float*)d_in[11];
  float* out = (float*)d_out;

  const size_t actBytes = (size_t)Bb * Hh * Ss * DPH * sizeof(_Float16);  // 16 MB
  const size_t wBytes   = (size_t)Dd * Dd * sizeof(_Float16);             // 2 MB
  char* w = (char*)d_ws;
  _Float16* q16   = (_Float16*)(w + 0 * actBytes);
  _Float16* k16   = (_Float16*)(w + 1 * actBytes);
  _Float16* v16   = (_Float16*)(w + 2 * actBytes);
  _Float16* ctx16 = (_Float16*)(w + 3 * actBytes);
  unsigned int* mbits = (unsigned int*)(w + 4 * actBytes);                // 2 MB
  char* wconv = w + 4 * actBytes + (size_t)Bb * Ss * (Ss / 32) * 4;
  _Float16* w16q = (_Float16*)(wconv + 0 * wBytes);
  _Float16* w16k = (_Float16*)(wconv + 1 * wBytes);
  _Float16* w16v = (_Float16*)(wconv + 2 * wBytes);
  _Float16* w16o = (_Float16*)(wconv + 3 * wBytes);

  // pre-passes
  cvt_f16_kernel<<<512, 256, 0, stream>>>(Wq, w16q);
  cvt_f16_kernel<<<512, 256, 0, stream>>>(Wk, w16k);
  cvt_f16_kernel<<<512, 256, 0, stream>>>(Wv, w16v);
  cvt_f16_kernel<<<512, 256, 0, stream>>>(Wo, w16o);
  mask_pack_kernel<<<2048, 256, 0, stream>>>(mask, mbits);

  const dim3 ggrid(Bb * Ss / 128, Dd / 64);
  const dim3 gblk(256);
  gemm_kernel<0><<<ggrid, gblk, 0, stream>>>(query, w16q, bq, q16, 0.125f);
  gemm_kernel<2><<<ggrid, gblk, 0, stream>>>(key,   w16k, bk, k16, 1.0f);
  gemm_kernel<0><<<ggrid, gblk, 0, stream>>>(value, w16v, bv, v16, 1.0f);

  const dim3 agrid(Ss / 128, Bb * Hh);
  attn_kernel<<<agrid, gblk, 0, stream>>>(q16, k16, v16, mbits, ctx16);

  gemm_kernel<3><<<ggrid, gblk, 0, stream>>>(ctx16, w16o, bo, out, 1.0f);
}
